// MultiheadAttention_66546223284288
// MI455X (gfx1250) — matrix operand order repaired and checked by reading, not compiled, not run on hardware
//
#include <hip/hip_runtime.h>
#include <hip/hip_bf16.h>

typedef _Float16 half4 __attribute__((ext_vector_type(4)));
typedef _Float16 half8 __attribute__((ext_vector_type(8)));
typedef _Float16 v16h  __attribute__((ext_vector_type(16)));
typedef float    v8f   __attribute__((ext_vector_type(8)));
typedef unsigned int u32x4 __attribute__((ext_vector_type(4)));
typedef int      i32x4 __attribute__((ext_vector_type(4)));
typedef int      i32x8 __attribute__((ext_vector_type(8)));

// ---------------------------------------------------------------------------
// WMMA fragment loaders per CDNA5 ISA 7.12.2 (wave32):
// A (16x32, MxK): lane(hi=L>>4, m=L&15): halves 0..7 = K hi*8+{0..7},
//                 halves 8..15 = K 16+hi*8+{0..7}, row m.
// B staged N-major: lane holds col n=L&15, K same as A (hi*8+{0..7}, 16+hi*8+{0..7}).
// C/D: lane L vgpr r <-> (row hi*8+r, col L&15).
// ---------------------------------------------------------------------------
__device__ __forceinline__ v16h ld_frag_a(const _Float16* base, int ld, int m0,
                                          int kb, int lane) {
  const int hi = (lane >> 4) & 1;
  const int mm = m0 + (lane & 15);
  const _Float16* p = base + mm * ld + kb + hi * 8;
  half8 lo = *(const half8*)p;
  half8 hv = *(const half8*)(p + 16);
  v16h r;
#pragma unroll
  for (int i = 0; i < 8; ++i) { r[i] = lo[i]; r[8 + i] = hv[i]; }
  return r;
}

__device__ __forceinline__ v16h ld_frag_b(const _Float16* base, int ld, int n0,
                                          int kb, int lane) {
  const int hi = (lane >> 4) & 1;
  const int nn = n0 + (lane & 15);
  const _Float16* p = base + nn * ld + kb + hi * 8;
  half8 lo = *(const half8*)p;
  half8 hv = *(const half8*)(p + 16);
  v16h r;
#pragma unroll
  for (int i = 0; i < 8; ++i) { r[i] = lo[i]; r[8 + i] = hv[i]; }
  return r;
}

__device__ __forceinline__ v8f wmma_f16(v16h a, v16h b, v8f c) {
  return __builtin_amdgcn_wmma_f32_16x16x32_f16(false, a, false, b, (short)0, c,
                                                false, false);
}

// ---------------------------------------------------------------------------
// Tensor Data Mover: DMA a 2D f16 tile (tile_d0 elems x tile_d1 rows, global
// row stride d0_stride elems) into LDS at lds_off, inserting pad_amount
// (code) DWORDs every (1<<pad_interval) 8-byte units (ISA ch.8 D# layout).
// 6-arg builtin form (clang-23 / therock-10.0 headers).
// ---------------------------------------------------------------------------
__device__ __forceinline__ void tdm_load_2d_f16(
    unsigned int lds_off, const _Float16* gptr,
    unsigned int tensor_d0, unsigned int tensor_d1, unsigned int d0_stride,
    unsigned int tile_d0, unsigned int tile_d1,
    unsigned int pad_interval_code, unsigned int pad_amount_code) {
  unsigned long long ga = (unsigned long long)(uintptr_t)gptr;
  u32x4 g0;
  g0[0] = 1u;                                    // count=1, user descriptor
  g0[1] = lds_off;                               // lds_addr (bytes)
  g0[2] = (unsigned int)(ga & 0xFFFFFFFFu);      // global_addr[31:0]
  g0[3] = (unsigned int)((ga >> 32) & 0x1FFFFFFu) | (2u << 30);  // type=2
  i32x8 g1;
  unsigned int w0 = (1u << 16);                  // data_size = 2 bytes
  w0 |= (1u << 20) | (pad_interval_code << 22) | (pad_amount_code << 25);
  g1[0] = (int)w0;
  g1[1] = (int)((tensor_d0 & 0xFFFFu) << 16);    // atomic_barrier=0 | dim0.lo
  g1[2] = (int)(((tensor_d0 >> 16) & 0xFFFFu) | ((tensor_d1 & 0xFFFFu) << 16));
  g1[3] = (int)(((tensor_d1 >> 16) & 0xFFFFu) | ((tile_d0 & 0xFFFFu) << 16));
  g1[4] = (int)(tile_d1 & 0xFFFFu);              // tile_dim1 | tile_dim2=0
  g1[5] = (int)d0_stride;                        // tensor_dim0_stride[31:0]
  g1[6] = 0;
  g1[7] = 0;
  i32x4 z4 = {0, 0, 0, 0};
  i32x8 z8 = {0, 0, 0, 0, 0, 0, 0, 0};
  __builtin_amdgcn_tensor_load_to_lds(g0, g1, z4, z4, z8, 0);
}

// ---------------------------------------------------------------------------
// fp32 -> f16 elementwise convert (float4 -> half4 per thread).
// ---------------------------------------------------------------------------
__global__ __launch_bounds__(256) void cvt_f32_f16(const float* __restrict__ in,
                                                   _Float16* __restrict__ out,
                                                   int n4) {
  int i = blockIdx.x * 256 + threadIdx.x;
  if (i < n4) {
    float4 v = ((const float4*)in)[i];
    half4 h;
    h[0] = (_Float16)v.x; h[1] = (_Float16)v.y;
    h[2] = (_Float16)v.z; h[3] = (_Float16)v.w;
    ((half4*)out)[i] = h;
  }
}

// ---------------------------------------------------------------------------
// GEMM: out = (X(4096x1024) @ W^T + bias) * outScale, all-f16 operands.
// Double-buffered TDM staging.  MODE 0: fp32 plain (B,S,D).
// MODE 1: f16 head-split (B,H,S,d_k).
// ---------------------------------------------------------------------------
#define GLDA 40  // 32 + 8 pad halves -> pad codes (3,3): 4 DW every 64B
template <int MODE>
__global__ __launch_bounds__(256) void gemm16_kernel(
    const _Float16* __restrict__ X, const _Float16* __restrict__ W,
    const float* __restrict__ bias, void* __restrict__ outv, float outScale) {
  __shared__ _Float16 As[2][128 * GLDA];
  __shared__ _Float16 Bs[2][64 * GLDA];

  const int t = threadIdx.x;
  const int lane = t & 31;
  const int w = t >> 5;
  const int wm = w >> 1;   // 0..3
  const int wn = w & 1;    // 0..1
  const int m0 = blockIdx.y * 128;
  const int n0 = blockIdx.x * 64;
  const int uw = __builtin_amdgcn_readfirstlane(w);

  v8f acc[2][2] = {};

  if (uw == 0) {
    tdm_load_2d_f16((unsigned int)(uintptr_t)&As[0][0], X + (size_t)m0 * 1024,
                    1024, 4096, 1024, 32, 128, 3, 3);
    tdm_load_2d_f16((unsigned int)(uintptr_t)&Bs[0][0], W + (size_t)n0 * 1024,
                    1024, 1024, 1024, 32, 64, 3, 3);
  }

  for (int i = 0; i < 32; ++i) {
    const int cur = i & 1;
    if (uw == 0) {
      if (i < 31) {
        const int kt = (i + 1) * 32;
        tdm_load_2d_f16((unsigned int)(uintptr_t)&As[cur ^ 1][0],
                        X + (size_t)m0 * 1024 + kt, 1024, 4096, 1024, 32, 128,
                        3, 3);
        tdm_load_2d_f16((unsigned int)(uintptr_t)&Bs[cur ^ 1][0],
                        W + (size_t)n0 * 1024 + kt, 1024, 1024, 1024, 32, 64,
                        3, 3);
        __builtin_amdgcn_s_wait_tensorcnt(2);  // tile i done, i+1 in flight
      } else {
        __builtin_amdgcn_s_wait_tensorcnt(0);
      }
    }
    __syncthreads();

    v16h af[2], bf[2];
#pragma unroll
    for (int mt = 0; mt < 2; ++mt)
      af[mt] = ld_frag_a(&As[cur][0], GLDA, wm * 32 + mt * 16, 0, lane);
#pragma unroll
    for (int nt = 0; nt < 2; ++nt)
      bf[nt] = ld_frag_b(&Bs[cur][0], GLDA, wn * 32 + nt * 16, 0, lane);
#pragma unroll
    for (int mt = 0; mt < 2; ++mt)
#pragma unroll
      for (int nt = 0; nt < 2; ++nt)
        acc[mt][nt] = wmma_f16(af[mt], bf[nt], acc[mt][nt]);
    __syncthreads();
  }

  const int hi = lane >> 4;
  const int nl = lane & 15;
  float bcol[2];
#pragma unroll
  for (int nt = 0; nt < 2; ++nt)
    bcol[nt] = bias[n0 + wn * 32 + nt * 16 + nl];

#pragma unroll
  for (int mt = 0; mt < 2; ++mt) {
#pragma unroll
    for (int nt = 0; nt < 2; ++nt) {
#pragma unroll
      for (int r = 0; r < 8; ++r) {
        int gm = m0 + wm * 32 + mt * 16 + hi * 8 + r;
        int gn = n0 + wn * 32 + nt * 16 + nl;
        float val = (acc[mt][nt][r] + bcol[nt]) * outScale;
        if (MODE == 0) {
          ((float*)outv)[(size_t)gm * 1024 + gn] = val;
        } else {
          int b = gm >> 11, s = gm & 2047, h = gn >> 6, d = gn & 63;
          ((_Float16*)outv)[(((size_t)(b * 16 + h)) * 2048 + s) * 64 + d] =
              (_Float16)val;
        }
      }
    }
  }
}

// ---------------------------------------------------------------------------
// Flash attention over f16 heads: 4 waves per 64 query rows of one (b,h),
// 32 key blocks of 64.  Q and K tiles staged by TDM; V staged transposed.
// ---------------------------------------------------------------------------
#define ALD 72  // 64 + 8 pad halves -> pad codes (4,3): 4 DW every 128B
__global__ __launch_bounds__(128) void attn_kernel(
    const _Float16* __restrict__ qh, const _Float16* __restrict__ kh,
    const _Float16* __restrict__ vh, const int* __restrict__ mask,
    _Float16* __restrict__ ctx) {
  __shared__ _Float16 Qs[64 * ALD];
  __shared__ _Float16 Ks[64 * ALD];
  __shared__ _Float16 Ps[64 * ALD];
  __shared__ _Float16 Vts[64 * ALD];
  __shared__ float Ss[64 * 68];
  __shared__ float rowm[64], rowl[64], rowscale[64];

  const int t = threadIdx.x;
  const int lane = t & 31;
  const int w = t >> 5;          // wave 0..3 owns query rows [w*16, w*16+16)
  const int bh = blockIdx.y;     // b*16 + h
  const int b = bh >> 4, h = bh & 15;
  const int q0 = blockIdx.x * 64;
  const size_t headOff = (size_t)bh * 2048 * 64;
  const int uw = __builtin_amdgcn_readfirstlane(w);
  const int hi = lane >> 4;
  const int nl = lane & 15;

  if (uw == 0) {   // Q tile (already scaled by 1/sqrt(D) in projection)
    tdm_load_2d_f16((unsigned int)(uintptr_t)&Qs[0],
                    qh + headOff + (size_t)q0 * 64, 64, 2048, 64, 64, 64, 4, 3);
  }
  if (t < 64) { rowm[t] = -1e30f; rowl[t] = 0.0f; }

  v8f o[4] = {};

  for (int j = 0; j < 32; ++j) {
    __syncthreads();
    if (uw == 0) {  // K block via TDM
      tdm_load_2d_f16((unsigned int)(uintptr_t)&Ks[0],
                      kh + headOff + (size_t)(j * 64) * 64, 64, 2048, 64, 64,
                      64, 4, 3);
    }
    // V block staged transposed (d-major) while the TDM runs.
#pragma unroll
    for (int i = 0; i < 4; ++i) {
      int lin = t + i * 128;     // 8 half8 per key row
      int kk = lin >> 3;
      int c8 = lin & 7;
      half8 vv = *(const half8*)(vh + headOff + (size_t)(j * 64 + kk) * 64 +
                                 c8 * 8);
#pragma unroll
      for (int e = 0; e < 8; ++e) Vts[(c8 * 8 + e) * ALD + kk] = vv[e];
    }
    if (uw == 0) __builtin_amdgcn_s_wait_tensorcnt(0);
    __syncthreads();

    // Scores: S = Q @ K^T.
    {
      v8f sacc[4] = {};
#pragma unroll
      for (int kb = 0; kb < 64; kb += 32) {
        v16h a = ld_frag_a(&Qs[0], ALD, w * 16, kb, lane);
#pragma unroll
        for (int nt = 0; nt < 4; ++nt) {
          v16h bf = ld_frag_b(&Ks[0], ALD, nt * 16, kb, lane);
          sacc[nt] = wmma_f16(a, bf, sacc[nt]);
        }
      }
#pragma unroll
      for (int nt = 0; nt < 4; ++nt)
#pragma unroll
        for (int r = 0; r < 8; ++r)
          Ss[(w * 16 + hi * 8 + r) * 68 + nt * 16 + nl] = sacc[nt][r];
    }
    __syncthreads();

    // Online softmax, one thread per query row; mask as int4.
    if (t < 64) {
      const int sq = q0 + t;
      const int4* mrow = (const int4*)(mask + (size_t)b * 2048 * 2048 +
                                       (size_t)sq * 2048 + j * 64);
      float mold = rowm[t];
      float rmax = -1e30f;
      for (int k4 = 0; k4 < 16; ++k4) {
        int4 mm = mrow[k4];
        float s0 = Ss[t * 68 + k4 * 4 + 0];
        float s1 = Ss[t * 68 + k4 * 4 + 1];
        float s2 = Ss[t * 68 + k4 * 4 + 2];
        float s3 = Ss[t * 68 + k4 * 4 + 3];
        if (mm.x == 0) s0 = -1e9f;
        if (mm.y == 0) s1 = -1e9f;
        if (mm.z == 0) s2 = -1e9f;
        if (mm.w == 0) s3 = -1e9f;
        Ss[t * 68 + k4 * 4 + 0] = s0;
        Ss[t * 68 + k4 * 4 + 1] = s1;
        Ss[t * 68 + k4 * 4 + 2] = s2;
        Ss[t * 68 + k4 * 4 + 3] = s3;
        rmax = fmaxf(rmax, fmaxf(fmaxf(s0, s1), fmaxf(s2, s3)));
      }
      float mnew = fmaxf(mold, rmax);
      float scale = __expf(mold - mnew);
      float psum = 0.0f;
      for (int kk = 0; kk < 64; ++kk) {
        float p = __expf(Ss[t * 68 + kk] - mnew);
        Ps[t * ALD + kk] = (_Float16)p;
        psum += p;
      }
      rowl[t] = rowl[t] * scale + psum;
      rowm[t] = mnew;
      rowscale[t] = scale;
    }
    __syncthreads();

    // Rescale O, then O += P @ V.
#pragma unroll
    for (int nt = 0; nt < 4; ++nt)
#pragma unroll
      for (int r = 0; r < 8; ++r)
        o[nt][r] *= rowscale[w * 16 + hi * 8 + r];
#pragma unroll
    for (int kb = 0; kb < 64; kb += 32) {
      v16h a = ld_frag_a(&Ps[0], ALD, w * 16, kb, lane);
#pragma unroll
      for (int nt = 0; nt < 4; ++nt) {
        v16h bf = ld_frag_b(&Vts[0], ALD, nt * 16, kb, lane);
        o[nt] = wmma_f16(a, bf, o[nt]);
      }
    }
  }

  // Normalize, write f16 merged (B,S,D) context.
#pragma unroll
  for (int nt = 0; nt < 4; ++nt) {
#pragma unroll
    for (int r = 0; r < 8; ++r) {
      int row = w * 16 + hi * 8 + r;
      float inv = 1.0f / rowl[row];
      int sq = q0 + row;
      int d = nt * 16 + nl;
      ctx[((size_t)(b * 2048 + sq)) * 1024 + h * 64 + d] =
          (_Float16)(o[nt][r] * inv);
    }
  }
}

// ---------------------------------------------------------------------------
extern "C" void kernel_launch(void* const* d_in, const int* in_sizes, int n_in,
                              void* d_out, int out_size, void* d_ws,
                              size_t ws_size, hipStream_t stream) {
  (void)in_sizes; (void)n_in; (void)out_size; (void)ws_size;
  const float* q    = (const float*)d_in[0];
  const float* k    = (const float*)d_in[1];
  const float* v    = (const float*)d_in[2];
  const int*   mask = (const int*)d_in[3];
  const float* Wq   = (const float*)d_in[4];
  const float* bq   = (const float*)d_in[5];
  const float* Wk   = (const float*)d_in[6];
  const float* bk   = (const float*)d_in[7];
  const float* Wv   = (const float*)d_in[8];
  const float* bv   = (const float*)d_in[9];
  const float* Wo   = (const float*)d_in[10];
  const float* bo   = (const float*)d_in[11];
  float* out = (float*)d_out;

  char* ws = (char*)d_ws;
  const size_t MB = 1u << 20;
  _Float16* Xq16 = (_Float16*)(ws + 0 * MB);
  _Float16* Xk16 = (_Float16*)(ws + 8 * MB);
  _Float16* Xv16 = (_Float16*)(ws + 16 * MB);
  _Float16* Wq16 = (_Float16*)(ws + 24 * MB);
  _Float16* Wk16 = (_Float16*)(ws + 26 * MB);
  _Float16* Wv16 = (_Float16*)(ws + 28 * MB);
  _Float16* Wo16 = (_Float16*)(ws + 30 * MB);
  _Float16* qh16 = (_Float16*)(ws + 32 * MB);
  _Float16* kh16 = (_Float16*)(ws + 40 * MB);
  _Float16* vh16 = (_Float16*)(ws + 48 * MB);
  _Float16* ctx16 = (_Float16*)(ws + 56 * MB);

  const int nAct4 = (2 * 2048 * 1024) / 4;   // activations, float4 count
  const int nW4 = (1024 * 1024) / 4;         // weights
  cvt_f32_f16<<<dim3((nAct4 + 255) / 256), dim3(256), 0, stream>>>(q, Xq16, nAct4);
  cvt_f32_f16<<<dim3((nAct4 + 255) / 256), dim3(256), 0, stream>>>(k, Xk16, nAct4);
  cvt_f32_f16<<<dim3((nAct4 + 255) / 256), dim3(256), 0, stream>>>(v, Xv16, nAct4);
  cvt_f32_f16<<<dim3((nW4 + 255) / 256), dim3(256), 0, stream>>>(Wq, Wq16, nW4);
  cvt_f32_f16<<<dim3((nW4 + 255) / 256), dim3(256), 0, stream>>>(Wk, Wk16, nW4);
  cvt_f32_f16<<<dim3((nW4 + 255) / 256), dim3(256), 0, stream>>>(Wv, Wv16, nW4);
  cvt_f32_f16<<<dim3((nW4 + 255) / 256), dim3(256), 0, stream>>>(Wo, Wo16, nW4);

  dim3 ggrid(16, 32), gblk(256);
  gemm16_kernel<1><<<ggrid, gblk, 0, stream>>>(Xq16, Wq16, bq, qh16, 0.03125f);
  gemm16_kernel<1><<<ggrid, gblk, 0, stream>>>(Xk16, Wk16, bk, kh16, 1.0f);
  gemm16_kernel<1><<<ggrid, gblk, 0, stream>>>(Xv16, Wv16, bv, vh16, 1.0f);
  attn_kernel<<<dim3(32, 32), dim3(128), 0, stream>>>(qh16, kh16, vh16, mask,
                                                      ctx16);
  gemm16_kernel<0><<<ggrid, gblk, 0, stream>>>(ctx16, Wo16, bo, out, 1.0f);
}
